// Prompt_Pool_17815524344308
// MI455X (gfx1250) — matrix operand order repaired and checked by reading, not compiled, not run on hardware
//
#include <hip/hip_runtime.h>
#include <hip/hip_bf16.h>
#include <math.h>

// ---------------- CDNA5 vector types ----------------
typedef __attribute__((ext_vector_type(16))) __bf16 v16bf;
typedef __attribute__((ext_vector_type(4)))  __bf16 v4bf;
typedef __attribute__((ext_vector_type(8)))  float  v8f;
typedef __attribute__((ext_vector_type(4)))  float  f32x4;

union Frag { v16bf v; f32x4 h[2]; };

// ---------------- problem constants ----------------
#define DIM   768
#define LSEQ  5
#define SEL   5
#define KC    128          // K-chunk staged in LDS (bf16)
#define STR   (KC + 8)     // LDS row stride in bf16 elems (bank spread)
#define ROWS  32           // rows per workgroup (2 waves x 16)
#define NTHR  64           // 2 wave32s
#define EPSN  1e-8f

// ---------------------------------------------------------------
// Kernel 1: reciprocal clipped row norms of x (B rows) and key (P)
// ---------------------------------------------------------------
__global__ void pp_norms(const float* __restrict__ x,
                         const float* __restrict__ pk,
                         float* __restrict__ rx,
                         float* __restrict__ rk, int B) {
    const int row = blockIdx.x;
    const int tid = threadIdx.x;
    const float* src;
    float* dst;
    if (row < B) { src = x  + (size_t)row * DIM;       dst = rx + row; }
    else         { src = pk + (size_t)(row - B) * DIM; dst = rk + (row - B); }

    float ss = 0.f;
    for (int d = tid; d < DIM; d += NTHR) { float v = src[d]; ss += v * v; }

    __shared__ float red[NTHR];
    red[tid] = ss;
    __syncthreads();
    #pragma unroll
    for (int off = NTHR / 2; off > 0; off >>= 1) {
        if (tid < off) red[tid] += red[tid + off];
        __syncthreads();
    }
    if (tid == 0) *dst = 1.0f / fmaxf(sqrtf(red[0]), EPSN);
}

// ---------------------------------------------------------------
// Kernel 2: fused  (bf16-WMMA cosine)  ->  top-5  ->  exact f32
//           re-score  ->  prompt gather
// One WG = 64 threads = 2 waves, owns 32 query rows.
// ---------------------------------------------------------------
__global__ void __launch_bounds__(NTHR)
pp_main(const float* __restrict__ x,
        const float* __restrict__ pk,
        const float* __restrict__ prompts,
        const float* __restrict__ rx,
        const float* __restrict__ rk,
        float* __restrict__ outDist,     // [B][SEL]
        float* __restrict__ outPrompt,   // [B][SEL][LSEQ*DIM]
        int B, int P) {
    __shared__ __bf16 Xs[ROWS * STR];          // 8704 B
    __shared__ __bf16 Ks[16 * STR];            // 4352 B
    __shared__ float  tileBuf[2][16][17];      // 2176 B
    __shared__ int    selIdx[ROWS][SEL];       //  640 B

    const int tid    = threadIdx.x;
    const int wave   = tid >> 5;
    const int lane   = tid & 31;
    const int laneLo = lane & 15;
    const int hi     = lane >> 4;              // 0 or 1
    const int rowBase = blockIdx.x * ROWS;
    const int wrow    = wave * 16;             // wave's row offset inside WG

    // running sorted top-5 (ascending) — meaningful on lanes 0..15 of each wave
    float bd[SEL] = {3.0e38f, 3.0e38f, 3.0e38f, 3.0e38f, 3.0e38f};
    int   bi[SEL] = {0, 0, 0, 0, 0};

    const int nCT = P >> 4;                    // 128 column tiles
    for (int ct = 0; ct < nCT; ++ct) {
        const int colBase = ct << 4;
        v8f acc = {};

        for (int kc = 0; kc < DIM; kc += KC) {
            __syncthreads();
            // ---- stage normalized X chunk (32 x 128 f32 -> bf16) ----
            #pragma unroll
            for (int i = 0; i < 16; ++i) {
                int linear = tid + NTHR * i;          // 0..1023
                int r = linear >> 5;                  // row 0..31
                int q = linear & 31;                  // float4 idx 0..31
                f32x4 v = *(const f32x4*)(x + (size_t)(rowBase + r) * DIM + kc + 4 * q);
                float s = rx[rowBase + r];
                v4bf h;
                h[0] = (__bf16)(v[0] * s); h[1] = (__bf16)(v[1] * s);
                h[2] = (__bf16)(v[2] * s); h[3] = (__bf16)(v[3] * s);
                *(v4bf*)&Xs[r * STR + 4 * q] = h;
            }
            // ---- stage normalized K tile (16 x 128) ----
            #pragma unroll
            for (int i = 0; i < 8; ++i) {
                int linear = tid + NTHR * i;          // 0..511
                int r = linear >> 5;                  // key row 0..15
                int q = linear & 31;
                f32x4 v = *(const f32x4*)(pk + (size_t)(colBase + r) * DIM + kc + 4 * q);
                float s = rk[colBase + r];
                v4bf h;
                h[0] = (__bf16)(v[0] * s); h[1] = (__bf16)(v[1] * s);
                h[2] = (__bf16)(v[2] * s); h[3] = (__bf16)(v[3] * s);
                *(v4bf*)&Ks[r * STR + 4 * q] = h;
            }
            __syncthreads();

            // ---- 4 WMMA k-steps (K = 32 each) ----
            const int am = wrow + laneLo;             // A row for this lane
            const int ac = hi ? 8 : 0;                // A K sub-offset
            const int bc = hi ? 8 : 0;                // B K sub-offset
            #pragma unroll
            for (int ks = 0; ks < KC / 32; ++ks) {
                const int k0 = ks * 32;
                Frag a, b;
                a.h[0] = *(const f32x4*)&Xs[am * STR + k0 + ac];
                a.h[1] = *(const f32x4*)&Xs[am * STR + k0 + ac + 16];
                b.h[0] = *(const f32x4*)&Ks[laneLo * STR + k0 + bc];
                b.h[1] = *(const f32x4*)&Ks[laneLo * STR + k0 + bc + 16];
                acc = __builtin_amdgcn_wmma_f32_16x16x32_bf16(
                        false, a.v, false, b.v, (short)0, acc, false, false);
            }
        }

        // ---- distances through LDS, per-row running top-5 ----
        #pragma unroll
        for (int r = 0; r < 8; ++r)
            tileBuf[wave][r + (hi ? 8 : 0)][laneLo] = 1.0f - acc[r];
        __syncthreads();

        if (lane < 16) {
            const int m = lane;
            #pragma unroll
            for (int n = 0; n < 16; ++n) {
                float v = tileBuf[wave][m][n];
                if (v < bd[SEL - 1]) {
                    bd[SEL - 1] = v; bi[SEL - 1] = colBase + n;
                    #pragma unroll
                    for (int j = SEL - 1; j > 0; --j) {
                        if (bd[j] < bd[j - 1]) {
                            float td = bd[j]; bd[j] = bd[j - 1]; bd[j - 1] = td;
                            int   ti = bi[j]; bi[j] = bi[j - 1]; bi[j - 1] = ti;
                        }
                    }
                }
            }
        }
        __syncthreads();
    }

    // ---- publish winning indices ----
    if (lane < 16) {
        #pragma unroll
        for (int j = 0; j < SEL; ++j) selIdx[wrow + lane][j] = bi[j];
    }
    __syncthreads();

    // ---- exact f32 re-score of the 5 winners per row ----
    for (int p = tid; p < ROWS * SEL; p += NTHR) {
        const int r = p / SEL, j = p % SEL;
        const int grow = rowBase + r;
        const int pidx = selIdx[r][j];
        const f32x4* xr = (const f32x4*)(x  + (size_t)grow * DIM);
        const f32x4* kr = (const f32x4*)(pk + (size_t)pidx * DIM);
        f32x4 s4 = {0.f, 0.f, 0.f, 0.f};
        #pragma unroll 4
        for (int d = 0; d < DIM / 4; ++d) s4 += xr[d] * kr[d];
        float dot = (s4[0] + s4[1]) + (s4[2] + s4[3]);
        outDist[(size_t)grow * SEL + j] = 1.0f - dot * rx[grow] * rk[pidx];
    }

    // ---- prompt gather: 32 rows x 5 sel x (LSEQ*DIM) floats ----
    const int ldrow = LSEQ * DIM;                     // 3840 floats
    for (int s = 0; s < ROWS * SEL; ++s) {
        const int r = s / SEL, j = s % SEL;
        const int grow = rowBase + r;
        const int pidx = selIdx[r][j];
        if (lane == 0 && s + 1 < ROWS * SEL) {
            const int nidx = selIdx[(s + 1) / SEL][(s + 1) % SEL];
            __builtin_prefetch(prompts + (size_t)nidx * ldrow, 0, 0);
        }
        const f32x4* src = (const f32x4*)(prompts + (size_t)pidx * ldrow);
        f32x4* dst = (f32x4*)(outPrompt + ((size_t)grow * SEL + j) * ldrow);
        for (int q = tid; q < ldrow / 4; q += NTHR) dst[q] = src[q];
    }
}

// ---------------------------------------------------------------
extern "C" void kernel_launch(void* const* d_in, const int* in_sizes, int n_in,
                              void* d_out, int out_size, void* d_ws, size_t ws_size,
                              hipStream_t stream) {
    const float* x  = (const float*)d_in[0];   // [B, 768]
    const float* pk = (const float*)d_in[1];   // [P, 768]
    const float* pr = (const float*)d_in[2];   // [P, 5, 768]
    const int B = in_sizes[0] / DIM;           // 8192
    const int P = in_sizes[1] / DIM;           // 2048

    float* rx = (float*)d_ws;                  // [B] reciprocal norms
    float* rk = rx + B;                        // [P]

    float* outDist   = (float*)d_out;          // [B,5]
    float* outPrompt = outDist + (size_t)B * SEL;

    pp_norms<<<B + P, NTHR, 0, stream>>>(x, pk, rx, rk, B);
    pp_main <<<B / ROWS, NTHR, 0, stream>>>(x, pk, pr, rx, rk,
                                            outDist, outPrompt, B, P);
}
